// CPI_GAT_6966436954190
// MI455X (gfx1250) — matrix operand order repaired and checked by reading, not compiled, not run on hardware
//
#include <hip/hip_runtime.h>

typedef __attribute__((ext_vector_type(16))) _Float16 v16h;
typedef __attribute__((ext_vector_type(8)))  _Float16 v8h;
typedef __attribute__((ext_vector_type(8)))  float    v8f;

#define CDIV(a,b) (((a)+(b)-1)/(b))

// ---------------- device helpers ----------------
__device__ __forceinline__ unsigned f2ord(float f) {
  unsigned b = __float_as_uint(f);
  return (b & 0x80000000u) ? ~b : (b | 0x80000000u);   // order-preserving map
}
__device__ __forceinline__ float ord2f(unsigned k) {
  unsigned b = (k & 0x80000000u) ? (k ^ 0x80000000u) : ~k;
  return __uint_as_float(b);
}
__device__ __forceinline__ float lrelu(float v) { return v >= 0.f ? v : 0.2f * v; }

// ---------------- fills / epilogues ----------------
__global__ __launch_bounds__(256) void k_fill_f32(float* p, float v, long long n) {
  long long t = (long long)blockIdx.x * blockDim.x + threadIdx.x;
  if (t < n) p[t] = v;
}
__global__ __launch_bounds__(256) void k_fill_u32(unsigned* p, unsigned v, long long n) {
  long long t = (long long)blockIdx.x * blockDim.x + threadIdx.x;
  if (t < n) p[t] = v;
}
__global__ __launch_bounds__(256) void k_init_bias(float* __restrict__ y, const float* __restrict__ b,
                                                   long long n, int F) {
  long long t = (long long)blockIdx.x * blockDim.x + threadIdx.x;
  if (t < n * F) y[t] = b[(int)(t % F)];
}
__global__ __launch_bounds__(256) void k_relu(float* y, long long n) {
  long long t = (long long)blockIdx.x * blockDim.x + threadIdx.x;
  if (t < n) y[t] = fmaxf(y[t], 0.f);
}

// ---------------- f16 staging for WMMA ----------------
// Xh: [n, Kpad] row-major f16, zero pad K>=Fin
__global__ __launch_bounds__(256) void k_cvt_x(const float* __restrict__ x, _Float16* __restrict__ xh,
                                               int n, int Fin, int Kpad) {
  long long t = (long long)blockIdx.x * blockDim.x + threadIdx.x;
  if (t >= (long long)n * Kpad) return;
  int i = (int)(t / Kpad), k = (int)(t % Kpad);
  xh[t] = (k < Fin) ? (_Float16)x[(long long)i * Fin + k] : (_Float16)0.f;
}
// Wt: transpose of W -> [Npad, Kpad] row-major f16 (row = output channel)
__global__ __launch_bounds__(256) void k_cvt_wt(const float* __restrict__ w, _Float16* __restrict__ wt,
                                                int Fin, int Fout, int Kpad, int Npad) {
  long long t = (long long)blockIdx.x * blockDim.x + threadIdx.x;
  if (t >= (long long)Npad * Kpad) return;
  int nn = (int)(t / Kpad), kk = (int)(t % Kpad);
  wt[t] = (nn < Fout && kk < Fin) ? (_Float16)w[(long long)kk * Fout + nn] : (_Float16)0.f;
}

// ---------------- WMMA GEMM: Y[n,Fout] = Xh[n,Kpad] * Wt[Npad,Kpad]^T ----------------
// One 16x16 output tile per wave (wave32), 8 waves per block.
// A frag (16x32 f16): lane L -> row = L&15; halves: K = (L>>4)*8 + 0..7 and 16+(L>>4)*8 + 0..7
// B frag (32x16 f16): lane L -> col = L&15; halves: K = (L>>4)*8 + 0..7 and 16+(L>>4)*8 + 0..7
// C/D f32: lane L, vgpr v -> row = (L>>4)*8 + v, col = L&15
__global__ __launch_bounds__(256) void k_gemm_f16_wmma(
    const _Float16* __restrict__ Xh, const _Float16* __restrict__ Wt,
    float* __restrict__ Y, int n, int Fout, int Kpad, int tilesN, int tiles)
{
  int wave = threadIdx.x >> 5;
  int tile = blockIdx.x * 8 + wave;
  if (tile >= tiles) return;
  int tm = tile / tilesN, tn = tile % tilesN;
  int lane = threadIdx.x & 31;
  int idx = lane & 15, half = lane >> 4;

  const _Float16* arow = Xh + (size_t)(tm * 16 + idx) * Kpad + half * 8;
  const _Float16* brow = Wt + (size_t)(tn * 16 + idx) * Kpad + half * 8;

  v8f acc = {};
  for (int k0 = 0; k0 < Kpad; k0 += 32) {
    v8h a0 = *(const v8h*)(arow + k0);
    v8h a1 = *(const v8h*)(arow + k0 + 16);
    v8h b0 = *(const v8h*)(brow + k0);
    v8h b1 = *(const v8h*)(brow + k0 + 16);
    if (k0 + 32 < Kpad) {
      __builtin_prefetch(arow + k0 + 32, 0, 3);   // global_prefetch_b8
      __builtin_prefetch(brow + k0 + 32, 0, 3);
    }
    v16h A, B;
#pragma unroll
    for (int i = 0; i < 8; ++i) { A[i] = a0[i]; A[i + 8] = a1[i]; B[i] = b0[i]; B[i + 8] = b1[i]; }
    acc = __builtin_amdgcn_wmma_f32_16x16x32_f16(false, A, false, B, (short)0, acc, false, false);
  }
  int col = tn * 16 + idx;
  if (col < Fout) {
    size_t base = (size_t)(tm * 16 + half * 8) * Fout + col;
#pragma unroll
    for (int v = 0; v < 8; ++v) Y[base + (size_t)v * Fout] = acc[v];
  }
}

// ---------------- GAT attention pipeline ----------------
// per-(node,head) attention dots: as = <h[i,hd,:], a_src[hd,:]>, ad likewise
__global__ __launch_bounds__(256) void k_node_alpha(
    const float* __restrict__ h, const float* __restrict__ a_src, const float* __restrict__ a_dst,
    float* __restrict__ as, float* __restrict__ ad, int n, int H, int C)
{
  long long t = (long long)blockIdx.x * blockDim.x + threadIdx.x;
  if (t >= (long long)n * H) return;
  int node = (int)(t / H), hd = (int)(t % H);
  const float* hp = h + (size_t)node * H * C + (size_t)hd * C;
  const float* s = a_src + (size_t)hd * C;
  const float* d = a_dst + (size_t)hd * C;
  float xs = 0.f, xd = 0.f;
  for (int c = 0; c < C; ++c) { float v = hp[c]; xs += v * s[c]; xd += v * d[c]; }
  as[t] = xs; ad[t] = xd;
}

// segment max over edges (incl. implicit self loops e in [E, E+n))
__global__ __launch_bounds__(256) void k_edge_max(
    const int* __restrict__ src, const int* __restrict__ dst,
    const float* __restrict__ as, const float* __restrict__ ad,
    unsigned* __restrict__ smax, int E, int n, int H)
{
  long long e = (long long)blockIdx.x * blockDim.x + threadIdx.x;
  if (e >= (long long)E + n) return;
  int s = (e < E) ? src[e] : (int)(e - E);
  int d = (e < E) ? dst[e] : (int)(e - E);
  for (int h = 0; h < H; ++h) {
    float v = lrelu(as[(size_t)s * H + h] + ad[(size_t)d * H + h]);
    atomicMax(&smax[(size_t)d * H + h], f2ord(v));
  }
}

// exp(e - max) and segment sum
__global__ __launch_bounds__(256) void k_edge_exp(
    const int* __restrict__ src, const int* __restrict__ dst,
    const float* __restrict__ as, const float* __restrict__ ad,
    const unsigned* __restrict__ smax, float* __restrict__ ssum,
    float* __restrict__ ex, int E, int n, int H)
{
  long long e = (long long)blockIdx.x * blockDim.x + threadIdx.x;
  if (e >= (long long)E + n) return;
  int s = (e < E) ? src[e] : (int)(e - E);
  int d = (e < E) ? dst[e] : (int)(e - E);
  for (int h = 0; h < H; ++h) {
    float v = lrelu(as[(size_t)s * H + h] + ad[(size_t)d * H + h]);
    float m = ord2f(smax[(size_t)d * H + h]);
    float x = __expf(v - m);
    ex[e * H + h] = x;
    atomicAdd(&ssum[(size_t)d * H + h], x);
  }
}

// wave-per-edge weighted aggregation: y[dst] += alpha * h[src]
__global__ __launch_bounds__(256) void k_edge_aggregate(
    const int* __restrict__ src, const int* __restrict__ dst,
    const float* __restrict__ ex, const float* __restrict__ ssum,
    const float* __restrict__ hfeat, float* __restrict__ y,
    int E, int n, int H, int C)
{
  long long t = (long long)blockIdx.x * blockDim.x + threadIdx.x;
  long long wid = t >> 5;
  int lane = (int)(t & 31);
  if (wid >= (long long)E + n) return;
  int s = (wid < E) ? src[wid] : (int)(wid - E);
  int d = (wid < E) ? dst[wid] : (int)(wid - E);
  int F = H * C;
  for (int hd = 0; hd < H; ++hd) {
    float coef = ex[wid * H + hd] / (ssum[(size_t)d * H + hd] + 1e-16f);
    const float* hp = hfeat + (size_t)s * F + (size_t)hd * C;
    float* yp = y + (size_t)d * F + (size_t)hd * C;
    for (int c = lane; c < C; c += 32) atomicAdd(yp + c, coef * hp[c]);
  }
}

// ---------------- GCN pipeline ----------------
__global__ __launch_bounds__(256) void k_deg_accum(const int* __restrict__ dst,
                                                   const float* __restrict__ w,
                                                   float* __restrict__ deg, int E) {
  long long t = (long long)blockIdx.x * blockDim.x + threadIdx.x;
  if (t < E) atomicAdd(&deg[dst[t]], w[t]);
}
__global__ __launch_bounds__(256) void k_dinv(const float* __restrict__ deg, float* __restrict__ dinv, int n) {
  long long t = (long long)blockIdx.x * blockDim.x + threadIdx.x;
  if (t < n) { float d = deg[t]; dinv[t] = d > 0.f ? rsqrtf(d) : 0.f; }
}
__global__ __launch_bounds__(256) void k_gcn_aggregate(
    const int* __restrict__ src, const int* __restrict__ dst,
    const float* __restrict__ w, const float* __restrict__ dinv,
    const float* __restrict__ hfeat, float* __restrict__ y, int E, int n, int C)
{
  long long t = (long long)blockIdx.x * blockDim.x + threadIdx.x;
  long long wid = t >> 5;
  int lane = (int)(t & 31);
  if (wid >= (long long)E + n) return;
  int s, d; float ew;
  if (wid < E) { s = src[wid]; d = dst[wid]; ew = w[wid]; }
  else         { s = d = (int)(wid - E); ew = 1.f; }
  float norm = dinv[s] * ew * dinv[d];
  const float* hp = hfeat + (size_t)s * C;
  float* yp = y + (size_t)d * C;
  for (int c = lane; c < C; c += 32) atomicAdd(yp + c, norm * hp[c]);
}

// ---------------- host orchestration ----------------
static void run_gat(hipStream_t st, const float* x, const int* src, const int* dst, int E,
                    const float* W, const float* a_src, const float* a_dst, const float* b,
                    int n, int Fin, int H, int C, float* y, bool relu,
                    _Float16* XH, _Float16* WT, float* HF, float* AS, float* AD,
                    unsigned* SMAX, float* SSUM, float* EX)
{
  int F = H * C;
  int Kpad = (Fin + 31) & ~31;
  int Npad = (F + 15) & ~15;
  int tilesM = CDIV(n, 16), tilesN = Npad / 16, tiles = tilesM * tilesN;

  k_cvt_x<<<(int)CDIV((long long)n * Kpad, 256), 256, 0, st>>>(x, XH, n, Fin, Kpad);
  k_cvt_wt<<<(int)CDIV((long long)Npad * Kpad, 256), 256, 0, st>>>(W, WT, Fin, F, Kpad, Npad);
  k_gemm_f16_wmma<<<CDIV(tiles, 8), 256, 0, st>>>(XH, WT, HF, n, F, Kpad, tilesN, tiles);
  k_node_alpha<<<(int)CDIV((long long)n * H, 256), 256, 0, st>>>(HF, a_src, a_dst, AS, AD, n, H, C);
  k_fill_u32<<<(int)CDIV((long long)n * H, 256), 256, 0, st>>>(SMAX, 0u, (long long)n * H);
  k_fill_f32<<<(int)CDIV((long long)n * H, 256), 256, 0, st>>>(SSUM, 0.f, (long long)n * H);
  k_init_bias<<<(int)CDIV((long long)n * F, 256), 256, 0, st>>>(y, b, (long long)n, F);
  long long Etot = (long long)E + n;
  k_edge_max<<<(int)CDIV(Etot, 256), 256, 0, st>>>(src, dst, AS, AD, SMAX, E, n, H);
  k_edge_exp<<<(int)CDIV(Etot, 256), 256, 0, st>>>(src, dst, AS, AD, SMAX, SSUM, EX, E, n, H);
  k_edge_aggregate<<<(int)CDIV(Etot * 32, 256), 256, 0, st>>>(src, dst, EX, SSUM, HF, y, E, n, H, C);
  if (relu) k_relu<<<(int)CDIV((long long)n * F, 256), 256, 0, st>>>(y, (long long)n * F);
}

static void run_gcn(hipStream_t st, const float* x, const int* src, const int* dst,
                    const float* ew, int E, const float* W, const float* b,
                    int n, int Fin, int Fout, float* y,
                    _Float16* XH, _Float16* WT, float* HF, float* DEG, float* DINV)
{
  k_fill_f32<<<(int)CDIV((long long)n, 256), 256, 0, st>>>(DEG, 1.f, (long long)n); // self-loop weight
  k_deg_accum<<<(int)CDIV((long long)E, 256), 256, 0, st>>>(dst, ew, DEG, E);
  k_dinv<<<(int)CDIV((long long)n, 256), 256, 0, st>>>(DEG, DINV, n);

  int Kpad = (Fin + 31) & ~31;
  int Npad = (Fout + 15) & ~15;
  int tilesM = CDIV(n, 16), tilesN = Npad / 16, tiles = tilesM * tilesN;
  k_cvt_x<<<(int)CDIV((long long)n * Kpad, 256), 256, 0, st>>>(x, XH, n, Fin, Kpad);
  k_cvt_wt<<<(int)CDIV((long long)Npad * Kpad, 256), 256, 0, st>>>(W, WT, Fin, Fout, Kpad, Npad);
  k_gemm_f16_wmma<<<CDIV(tiles, 8), 256, 0, st>>>(XH, WT, HF, n, Fout, Kpad, tilesN, tiles);

  k_init_bias<<<(int)CDIV((long long)n * Fout, 256), 256, 0, st>>>(y, b, (long long)n, Fout);
  long long Etot = (long long)E + n;
  k_gcn_aggregate<<<(int)CDIV(Etot * 32, 256), 256, 0, st>>>(src, dst, ew, DINV, HF, y, E, n, Fout);
  k_relu<<<(int)CDIV((long long)n * Fout, 256), 256, 0, st>>>(y, (long long)n * Fout);
}

extern "C" void kernel_launch(void* const* d_in, const int* in_sizes, int n_in,
                              void* d_out, int out_size, void* d_ws, size_t ws_size,
                              hipStream_t stream)
{
  (void)n_in; (void)out_size; (void)ws_size;
  const float* mol_x  = (const float*)d_in[0];
  const int*   mol_ei = (const int*)d_in[1];
  const float* pro_x  = (const float*)d_in[2];
  const int*   pro_ei = (const int*)d_in[3];
  const float* pro_ew = (const float*)d_in[4];
  const float *W1 = (const float*)d_in[5],  *as1 = (const float*)d_in[6],
              *ad1 = (const float*)d_in[7], *b1  = (const float*)d_in[8];
  const float *W2 = (const float*)d_in[9],  *as2 = (const float*)d_in[10],
              *ad2 = (const float*)d_in[11],*b2  = (const float*)d_in[12];
  const float *W3 = (const float*)d_in[13], *as3 = (const float*)d_in[14],
              *ad3 = (const float*)d_in[15],*b3  = (const float*)d_in[16];
  const float *Wp1 = (const float*)d_in[17], *bp1 = (const float*)d_in[18];
  const float *Wp2 = (const float*)d_in[19], *aps2 = (const float*)d_in[20],
              *apd2 = (const float*)d_in[21],*bp2 = (const float*)d_in[22];
  const float *Wp3 = (const float*)d_in[23], *aps3 = (const float*)d_in[24],
              *apd3 = (const float*)d_in[25],*bp3 = (const float*)d_in[26];

  const int n_mol = in_sizes[0] / 78;
  const int n_pro = in_sizes[2] / 33;
  const int E_mol = in_sizes[1] / 2;
  const int E_pro = in_sizes[3] / 2;
  const int* msrc = mol_ei;  const int* mdst = mol_ei + E_mol;
  const int* psrc = pro_ei;  const int* pdst = pro_ei + E_pro;

  // ---- workspace arena (bump alloc, 256B aligned) ----
  char* base = (char*)d_ws;
  size_t off = 0;
  auto take = [&](size_t bytes) -> void* {
    void* p = base + off;
    off += (bytes + 255) & ~(size_t)255;
    return p;
  };
  size_t xh_b = (size_t)n_mol * 320 * 2;                    // mol Kpad max=320
  size_t xh_p = (size_t)n_pro * 160 * 2;                    // pro Kpad max=160
  _Float16* XH = (_Float16*)take(xh_b > xh_p ? xh_b : xh_p);
  _Float16* WT = (_Float16*)take((size_t)320 * 320 * 2);
  size_t hf_b = (size_t)n_mol * 312 * 4, hf_p = (size_t)n_pro * 132 * 4;
  size_t hf_sz = hf_b > hf_p ? hf_b : hf_p;
  float* HF = (float*)take(hf_sz);
  float* Y1 = (float*)take(hf_sz);
  float* Y2 = (float*)take(hf_sz);
  size_t nmax = (size_t)(n_mol > n_pro ? n_mol : n_pro);
  float*    AS   = (float*)take(nmax * 2 * 4);
  float*    AD   = (float*)take(nmax * 2 * 4);
  unsigned* SMAX = (unsigned*)take(nmax * 2 * 4);
  float*    SSUM = (float*)take(nmax * 2 * 4);
  size_t etot_m = (size_t)E_mol + n_mol, etot_p = (size_t)E_pro + n_pro;
  size_t emax = etot_m > etot_p ? etot_m : etot_p;
  float* EX = (float*)take(emax * 2 * 4);

  float* out_mol = (float*)d_out;                       // [n_mol, 312]
  float* out_pro = out_mol + (size_t)n_mol * 312;       // [n_pro, 132]

  // ---- mol branch: GAT(78->2x78) relu, GAT(156->2x156) relu, GAT(312->1x312) ----
  run_gat(stream, mol_x, msrc, mdst, E_mol, W1, as1, ad1, b1, n_mol, 78, 2, 78,  Y1, true,
          XH, WT, HF, AS, AD, SMAX, SSUM, EX);
  run_gat(stream, Y1,    msrc, mdst, E_mol, W2, as2, ad2, b2, n_mol, 156, 2, 156, Y2, true,
          XH, WT, HF, AS, AD, SMAX, SSUM, EX);
  run_gat(stream, Y2,    msrc, mdst, E_mol, W3, as3, ad3, b3, n_mol, 312, 1, 312, out_mol, false,
          XH, WT, HF, AS, AD, SMAX, SSUM, EX);

  // ---- pro branch: GCN(33->33) relu, GAT(33->2x66) relu, GAT(132->1x132) relu ----
  run_gcn(stream, pro_x, psrc, pdst, pro_ew, E_pro, Wp1, bp1, n_pro, 33, 33, Y1,
          XH, WT, HF, AS, AD);   // AS/AD reused as deg/dinv
  run_gat(stream, Y1, psrc, pdst, E_pro, Wp2, aps2, apd2, bp2, n_pro, 33, 2, 66,  Y2, true,
          XH, WT, HF, AS, AD, SMAX, SSUM, EX);
  run_gat(stream, Y2, psrc, pdst, E_pro, Wp3, aps3, apd3, bp3, n_pro, 132, 1, 132, out_pro, true,
          XH, WT, HF, AS, AD, SMAX, SSUM, EX);
}
